// TemporalAttentionModule_9680856285923
// MI455X (gfx1250) — matrix operand order repaired and checked by reading, not compiled, not run on hardware
//
#include <hip/hip_runtime.h>
#include <hip/hip_bf16.h>
#include <math.h>

// Problem constants (match reference)
#define BB 8
#define FF 128
#define KK 256
#define EE 256
#define ALPHA 0.2f
#define TI 8          // i-rows per block in the score kernel

typedef __attribute__((ext_vector_type(16))) _Float16 v16h;
typedef __attribute__((ext_vector_type(8)))  float    v8f;

// ---------------------------------------------------------------------------
// Kernel 1: per-batch GEMMs  pp = v @ W[:F] + b_lin,  qq = v @ W[F:]
//   v[b,k,f] = x[b,f,k]   (x is [B,F,K])
// One wave computes one 16x16 tile of BOTH pp and qq sharing the A fragment.
// Per-lane gathers are strided but coalesce across lanes (lm varies fastest).
// ---------------------------------------------------------------------------
__global__ void tam_proj_wmma(const float* __restrict__ x,
                              const float* __restrict__ W,
                              const float* __restrict__ b_lin,
                              float* __restrict__ pp,
                              float* __restrict__ qq)
{
    const int lane = threadIdx.x;      // 0..31 (wave32)
    const int lm   = lane & 15;
    const int hi   = lane >> 4;        // lane group 0/1
    const int n0   = blockIdx.x * 16;  // E tile
    const int m0   = blockIdx.y * 16;  // K (node) tile
    const int b    = blockIdx.z;

    v8f accP = {};
    v8f accQ = {};

    for (int k0 = 0; k0 < FF; k0 += 32) {
        // A fragment: A[m][k] = x[b, k0+k, m0+m]; m = lm
        v16h afrag;
#pragma unroll
        for (int e = 0; e < 16; ++e) {
            const int k = 16 * hi + e;
            afrag[e] = (_Float16)x[((size_t)b * FF + (k0 + k)) * KK + (m0 + lm)];
        }
        // B fragments: Btop[k][n] = W[k0+k, n0+n], Bbot rows +F; n = lm, k = 16*hi+e
        v16h btop, bbot;
#pragma unroll
        for (int e = 0; e < 16; ++e) {
            const int k = 16 * hi + e;
            btop[e] = (_Float16)W[(size_t)(k0 + k) * EE + (n0 + lm)];
            bbot[e] = (_Float16)W[(size_t)(k0 + k + FF) * EE + (n0 + lm)];
        }
        accP = __builtin_amdgcn_wmma_f32_16x16x32_f16(false, afrag, false, btop,
                                                      (short)0, accP, false, false);
        accQ = __builtin_amdgcn_wmma_f32_16x16x32_f16(false, afrag, false, bbot,
                                                      (short)0, accQ, false, false);
    }

    const float bl = b_lin[n0 + lm];
#pragma unroll
    for (int r = 0; r < 8; ++r) {
        const int m = r + 8 * hi;                    // C/D layout: m = r + 8*hi, n = lm
        const size_t idx = ((size_t)b * KK + (m0 + m)) * EE + (n0 + lm);
        pp[idx] = accP[r] + bl;                      // fold lin bias into p
        qq[idx] = accQ[r];
    }
}

// ---------------------------------------------------------------------------
// Kernel 2: pairwise scores + softmax, i-tiled to cut L2 traffic 8x.
//   e[b,i,j] = sum_e lrelu(pp[b,i,e] + qq[b,j,e]) * a[e] + bias[i,j]
// One 256-thread block per (b, i-tile of TI=8 rows); thread t == j.
// Each thread streams its own q row ONCE (float4) and updates TI accumulators;
// pp rows + a live in LDS and are read as float4 (ds_load_b128).
// ---------------------------------------------------------------------------
__global__ void tam_score_softmax(const float* __restrict__ pp,
                                  const float* __restrict__ qq,
                                  const float* __restrict__ a,
                                  const float* __restrict__ bias,
                                  float* __restrict__ attn)
{
    __shared__ __align__(16) float pp_s[TI][EE];
    __shared__ __align__(16) float a_s[EE];
    __shared__ float red[8];

    const int j  = threadIdx.x;            // 0..255
    const int i0 = (blockIdx.x & 31) * TI; // K/TI == 32 tiles per batch
    const int b  = blockIdx.x >> 5;

    // TI consecutive pp rows are contiguous in memory: coalesced bulk load.
#pragma unroll
    for (int t = 0; t < TI; ++t)
        ((float*)pp_s)[t * KK + j] = pp[((size_t)b * KK + i0) * EE + t * KK + j];
    a_s[j] = a[j];
    __syncthreads();

    const float* qrow = qq + ((size_t)b * KK + j) * EE;

    float acc[TI];
#pragma unroll
    for (int ii = 0; ii < TI; ++ii) acc[ii] = 0.0f;

    for (int e = 0; e < EE; e += 4) {
        const float4 qv = *(const float4*)(qrow + e);
        const float4 av = *(const float4*)(&a_s[e]);
#pragma unroll
        for (int ii = 0; ii < TI; ++ii) {
            const float4 pv = *(const float4*)(&pp_s[ii][e]);
            float t0 = pv.x + qv.x; t0 = (t0 > 0.0f) ? t0 : ALPHA * t0;
            float t1 = pv.y + qv.y; t1 = (t1 > 0.0f) ? t1 : ALPHA * t1;
            float t2 = pv.z + qv.z; t2 = (t2 > 0.0f) ? t2 : ALPHA * t2;
            float t3 = pv.w + qv.w; t3 = (t3 > 0.0f) ? t3 : ALPHA * t3;
            float s  = fmaf(t0, av.x, acc[ii]);
            s        = fmaf(t1, av.y, s);
            s        = fmaf(t2, av.z, s);
            acc[ii]  = fmaf(t3, av.w, s);
        }
    }

    // ---- per-row softmax over j (wave32 shuffles + LDS across 8 waves) ----
#pragma unroll 1
    for (int ii = 0; ii < TI; ++ii) {
        const int i = i0 + ii;
        const float ej = acc[ii] + bias[(size_t)i * KK + j];

        float m = ej;
#pragma unroll
        for (int off = 16; off > 0; off >>= 1)
            m = fmaxf(m, __shfl_xor(m, off, 32));
        if ((j & 31) == 0) red[j >> 5] = m;
        __syncthreads();
        float gm = red[0];
#pragma unroll
        for (int w = 1; w < 8; ++w) gm = fmaxf(gm, red[w]);
        __syncthreads();                       // red reads done before reuse

        const float p = __expf(ej - gm);
        float ssum = p;
#pragma unroll
        for (int off = 16; off > 0; off >>= 1)
            ssum += __shfl_xor(ssum, off, 32);
        if ((j & 31) == 0) red[j >> 5] = ssum;
        __syncthreads();
        float gs = red[0];
#pragma unroll
        for (int w = 1; w < 8; ++w) gs += red[w];
        __syncthreads();                       // red reads done before next ii

        attn[((size_t)b * KK + i) * KK + j] = p / gs;
    }
}

// ---------------------------------------------------------------------------
// Kernel 3: h_time = sigmoid(attn @ v); output stored transposed: out[b,f,i].
//   sum_j attn[b,i,j] * x[b,f,j]
// All per-lane fragment reads and the transposed store are contiguous ->
// float4 (global_load_b128 / global_store_b128).
// ---------------------------------------------------------------------------
__global__ void tam_out_wmma(const float* __restrict__ attn,
                             const float* __restrict__ x,
                             float* __restrict__ out)
{
    const int lane = threadIdx.x;
    const int lm   = lane & 15;
    const int hi   = lane >> 4;
    const int n0   = blockIdx.x * 16;  // F tile (f)
    const int m0   = blockIdx.y * 16;  // K tile (i)
    const int b    = blockIdx.z;

    v8f acc = {};
    for (int k0 = 0; k0 < KK; k0 += 32) {
        // A[m][k] = attn[b, m0+m, k0+k]; m = lm. k = 16*hi + e: 16 contiguous floats.
        const float* arow = attn + ((size_t)b * KK + (m0 + lm)) * KK + k0 + 16 * hi;
        const float4 a0 = *(const float4*)(arow + 0);
        const float4 a1 = *(const float4*)(arow + 4);
        const float4 a2 = *(const float4*)(arow + 8);
        const float4 a3 = *(const float4*)(arow + 12);
        v16h afrag;
        afrag[0]  = (_Float16)a0.x; afrag[1]  = (_Float16)a0.y;
        afrag[2]  = (_Float16)a0.z; afrag[3]  = (_Float16)a0.w;
        afrag[4]  = (_Float16)a1.x; afrag[5]  = (_Float16)a1.y;
        afrag[6]  = (_Float16)a1.z; afrag[7]  = (_Float16)a1.w;
        afrag[8]  = (_Float16)a2.x; afrag[9]  = (_Float16)a2.y;
        afrag[10] = (_Float16)a2.z; afrag[11] = (_Float16)a2.w;
        afrag[12] = (_Float16)a3.x; afrag[13] = (_Float16)a3.y;
        afrag[14] = (_Float16)a3.z; afrag[15] = (_Float16)a3.w;

        // B[k][n] = x[b, n0+n, k0+k]; n = lm, k = 16*hi + e: 16 contiguous floats
        const float* brow = x + ((size_t)b * FF + (n0 + lm)) * KK + k0 + 16 * hi;
        const float4 b0 = *(const float4*)(brow + 0);
        const float4 b1 = *(const float4*)(brow + 4);
        const float4 b2 = *(const float4*)(brow + 8);
        const float4 b3 = *(const float4*)(brow + 12);
        v16h bfrag;
        bfrag[0]  = (_Float16)b0.x; bfrag[1]  = (_Float16)b0.y;
        bfrag[2]  = (_Float16)b0.z; bfrag[3]  = (_Float16)b0.w;
        bfrag[4]  = (_Float16)b1.x; bfrag[5]  = (_Float16)b1.y;
        bfrag[6]  = (_Float16)b1.z; bfrag[7]  = (_Float16)b1.w;
        bfrag[8]  = (_Float16)b2.x; bfrag[9]  = (_Float16)b2.y;
        bfrag[10] = (_Float16)b2.z; bfrag[11] = (_Float16)b2.w;
        bfrag[12] = (_Float16)b3.x; bfrag[13] = (_Float16)b3.y;
        bfrag[14] = (_Float16)b3.z; bfrag[15] = (_Float16)b3.w;

        acc = __builtin_amdgcn_wmma_f32_16x16x32_f16(false, afrag, false, bfrag,
                                                     (short)0, acc, false, false);
    }

    // out[b, f=n0+lm, i=m0+m], m = r + 8*hi: 8 contiguous floats per lane.
    float4 o0, o1;
    o0.x = 1.0f / (1.0f + __expf(-acc[0]));
    o0.y = 1.0f / (1.0f + __expf(-acc[1]));
    o0.z = 1.0f / (1.0f + __expf(-acc[2]));
    o0.w = 1.0f / (1.0f + __expf(-acc[3]));
    o1.x = 1.0f / (1.0f + __expf(-acc[4]));
    o1.y = 1.0f / (1.0f + __expf(-acc[5]));
    o1.z = 1.0f / (1.0f + __expf(-acc[6]));
    o1.w = 1.0f / (1.0f + __expf(-acc[7]));
    float* orow = out + ((size_t)b * FF + (n0 + lm)) * KK + m0 + 8 * hi;
    *(float4*)(orow + 0) = o0;
    *(float4*)(orow + 4) = o1;
}

// ---------------------------------------------------------------------------
extern "C" void kernel_launch(void* const* d_in, const int* in_sizes, int n_in,
                              void* d_out, int out_size, void* d_ws, size_t ws_size,
                              hipStream_t stream)
{
    (void)in_sizes; (void)n_in; (void)out_size; (void)ws_size;

    const float* x     = (const float*)d_in[0];  // [B,F,K]
    const float* W     = (const float*)d_in[1];  // [2F,E]
    const float* b_lin = (const float*)d_in[2];  // [E]
    const float* a     = (const float*)d_in[3];  // [E]
    const float* bias  = (const float*)d_in[4];  // [K,K]
    float*       out   = (float*)d_out;          // [B,F,K]

    // Workspace layout (floats): pp | qq | attn  (3 * B*K*E = 6 MB)
    float* pp   = (float*)d_ws;
    float* qq   = pp + (size_t)BB * KK * EE;
    float* attn = qq + (size_t)BB * KK * EE;

    // k1: one wave per 16x16 tile of (K x E), all batches
    dim3 g1(EE / 16, KK / 16, BB);
    tam_proj_wmma<<<g1, dim3(32, 1, 1), 0, stream>>>(x, W, b_lin, pp, qq);

    // k2: one 256-thread block per (b, TI-row i-tile)
    tam_score_softmax<<<dim3(BB * (KK / TI), 1, 1), dim3(KK, 1, 1), 0, stream>>>(
        pp, qq, a, bias, attn);

    // k3: one wave per 16x16 tile of (K x F), all batches
    dim3 g3(FF / 16, KK / 16, BB);
    tam_out_wmma<<<g3, dim3(32, 1, 1), 0, stream>>>(attn, x, out);
}
